// EulerCausalAttention_75892072121064
// MI455X (gfx1250) — matrix operand order repaired and checked by reading, not compiled, not run on hardware
//
#include <hip/hip_runtime.h>
#include <hip/hip_bf16.h>
#include <math.h>

#define DMODEL 1024
#define NHEADS 16
#define DHEAD  64
#define EDIM   128   // Q/K feature dim = 2*DHEAD (cos||sin)
#define BATCH  2
#define SEQ    2048
#define LUTN   4096

typedef __attribute__((ext_vector_type(16))) _Float16 v16h;
typedef __attribute__((ext_vector_type(8)))  _Float16 v8h;
typedef __attribute__((ext_vector_type(8)))  float    v8f;

// ---------------------------------------------------------------------------
// WMMA helpers (gfx1250 wave32, 16x16x32 f16 -> f32)
// ---------------------------------------------------------------------------
__device__ __forceinline__ v8f wmma16(v16h a, v16h b, v8f c) {
  // (neg_a, A, neg_b, B, c_mod, C, reuse_a, reuse_b)
  return __builtin_amdgcn_wmma_f32_16x16x32_f16(false, a, false, b, (short)0, c,
                                                false, false);
}

// A-matrix f16 16x32 fragment: lane L holds row M=L%16;
// lanes 0-15: K k0..k0+15; lanes 16-31: k0+16..k0+31.
__device__ __forceinline__ v16h load_a_frag(const _Float16* __restrict__ A, int lda,
                                            int m0, int k0, int lane) {
  const int row = m0 + (lane & 15);
  const int kb  = k0 + ((lane & 16) ? 16 : 0);
  const _Float16* p = A + (size_t)row * lda + kb;
  v8h lo = *(const v8h*)(p);
  v8h hi = *(const v8h*)(p + 8);
  return __builtin_shufflevector(lo, hi, 0,1,2,3,4,5,6,7,8,9,10,11,12,13,14,15);
}

// B-matrix f16 32x16 fragment. B provided as N x K row-major (i.e. B^T),
// so a fragment is 16 contiguous halves: lane n -> column N=n0+n%16,
// K range k0..k0+15 (lanes 0-15) / k0+16..k0+31 (lanes 16-31).
__device__ __forceinline__ v16h load_b_frag(const _Float16* __restrict__ B, int ldb,
                                            int n0, int k0, int lane) {
  const int row = n0 + (lane & 15);
  const int kb  = k0 + ((lane & 16) ? 16 : 0);
  return *(const v16h*)(B + (size_t)row * ldb + kb);
}

// ---------------------------------------------------------------------------
// Raw 32-bit LDS offset of a __shared__ object (needed as VDST of the
// gfx1250 async-to-LDS instructions, whose LDS address comes from VDST plus offset)
// ---------------------------------------------------------------------------
typedef __attribute__((address_space(3))) _Float16 lds_half_t;
__device__ __forceinline__ unsigned lds_off_of(const _Float16* p) {
  return (unsigned)(unsigned long long)(const lds_half_t*)p;
}

// One 16-byte async global->LDS copy (tracked by ASYNCcnt).
__device__ __forceinline__ void async_copy_b128(unsigned lds_off, const void* gaddr) {
  asm volatile("global_load_async_to_lds_b128 %0, %1, off"
               :: "v"(lds_off), "v"(gaddr) : "memory");
}

// Stage one 64(rows) x 32(k) f16 B tile into LDS: 4 KB, 128 threads x 32 B.
// LDS layout: halves[row*32 + k].
__device__ __forceinline__ void stage_b_tile(const _Float16* __restrict__ Bw,
                                             unsigned ldsbase, int n0, int k0, int tid) {
  const int sr = tid >> 1;   // 0..63 : row within tile
  const int sh = tid & 1;    // 0..1  : 16-half chunk
  const _Float16* src = Bw + (size_t)(n0 + sr) * DMODEL + k0 + sh * 16;
  const unsigned dst = ldsbase + (unsigned)(sr * 64 + sh * 32);
  async_copy_b128(dst,       src);
  async_copy_b128(dst + 16u, src + 8);
}

// One k-step: 2 A fragments (32 M rows) x 4 B sub-tiles from LDS -> 8 WMMA.
__device__ __forceinline__ void compute_step(const _Float16* __restrict__ A,
                                             const _Float16* Bs,  // shared buffer
                                             int m0, int k0, int lane, v8f* acc) {
  v16h a0 = load_a_frag(A, DMODEL, m0,      k0, lane);
  v16h a1 = load_a_frag(A, DMODEL, m0 + 16, k0, lane);
  const int kc = (lane & 16) ? 16 : 0;
#pragma unroll
  for (int j = 0; j < 4; ++j) {
    v16h bm = *(const v16h*)(Bs + (j * 16 + (lane & 15)) * 32 + kc);
    acc[j]     = wmma16(a0, bm, acc[j]);
    acc[4 + j] = wmma16(a1, bm, acc[4 + j]);
  }
}

// ---------------------------------------------------------------------------
// Shared GEMM core: block = 128 threads (4 waves), C tile = 128x64
// (each wave: 32 M rows x 64 N cols, acc[8] fragments).
// B tile double-buffered in LDS via async copies, manually unrolled so the
// buffer parity is static (avoids acc phi-copies + hazard NOPs), one barrier
// per k-step:  wait(own asyncs) -> barrier -> stage(next) -> compute(cur).
// ---------------------------------------------------------------------------
__device__ __forceinline__ void gemm_core(const _Float16* __restrict__ A,
                                          const _Float16* __restrict__ Bw,
                                          _Float16* Bt,   // shared: 2*64*32 halves
                                          int m0, int n0, int tid, v8f* acc) {
  const int lane = tid & 31;
  const unsigned lb0 = lds_off_of(Bt);
  const unsigned lb1 = lb0 + 64u * 32u * 2u;
  const _Float16* Bt0 = Bt;
  const _Float16* Bt1 = Bt + 64 * 32;

  stage_b_tile(Bw, lb0, n0, 0, tid);
  int k = 0;
#pragma unroll 1
  for (int ii = 0; ii < 15; ++ii) {          // steps 0..29 (pairs)
    asm volatile("s_wait_asynccnt 0x0" ::: "memory");
    __syncthreads();
    stage_b_tile(Bw, lb1, n0, k + 32, tid);
    compute_step(A, Bt0, m0, k, lane, acc);

    asm volatile("s_wait_asynccnt 0x0" ::: "memory");
    __syncthreads();
    stage_b_tile(Bw, lb0, n0, k + 64, tid);
    compute_step(A, Bt1, m0, k + 32, lane, acc);
    k += 64;
  }
  // tail: steps 30 (k=960) and 31 (k=992)
  asm volatile("s_wait_asynccnt 0x0" ::: "memory");
  __syncthreads();
  stage_b_tile(Bw, lb1, n0, k + 32, tid);
  compute_step(A, Bt0, m0, k, lane, acc);

  asm volatile("s_wait_asynccnt 0x0" ::: "memory");
  __syncthreads();
  compute_step(A, Bt1, m0, k + 32, lane, acc);
}

// ---------------------------------------------------------------------------
// Kernel 0: f32 -> f16 weight conversion (v_w, out_w)
// ---------------------------------------------------------------------------
__global__ void f32_to_f16_kernel(const float* __restrict__ src,
                                  _Float16* __restrict__ dst) {
  const size_t i = (size_t)blockIdx.x * blockDim.x + threadIdx.x;
  dst[i] = (_Float16)src[i];
}

// ---------------------------------------------------------------------------
// Kernel 1: Euler features. x -> Xh (f16), Q/K (B,H,S,128) f16 = [cos||sin]
// LUT semantics: idx = round(theta*LUTN/2pi) mod LUTN ; angle = idx*2pi/LUTN
// ---------------------------------------------------------------------------
__global__ void qk_prep_kernel(const float* __restrict__ x,
                               const float* __restrict__ w_q, const float* __restrict__ b_q,
                               const float* __restrict__ w_k, const float* __restrict__ b_k,
                               _Float16* __restrict__ Xh,
                               _Float16* __restrict__ Qh, _Float16* __restrict__ Kh) {
  const size_t i = (size_t)blockIdx.x * blockDim.x + threadIdx.x;  // < B*S*D
  const float xv = x[i];
  Xh[i] = (_Float16)xv;

  const int d  = (int)(i & (DMODEL - 1));
  const int s  = (int)((i >> 10) & (SEQ - 1));
  const int b  = (int)(i >> 21);
  const int h  = d >> 6;
  const int dh = d & 63;

  const float k2i = 651.8986469f;        // LUTN / (2*pi)
  const float i2a = 1.5339807879e-3f;    // (2*pi) / LUTN

  float tq = xv / (1.0f + fabsf(w_q[d])) + b_q[d];
  int   iq = ((int)rintf(tq * k2i)) & (LUTN - 1);
  float sq, cq; __sincosf((float)iq * i2a, &sq, &cq);

  float tk = xv / (1.0f + fabsf(w_k[d])) + b_k[d];
  int   ik = ((int)rintf(tk * k2i)) & (LUTN - 1);
  float sk, ck; __sincosf((float)ik * i2a, &sk, &ck);

  const size_t base = (((size_t)b * NHEADS + h) * SEQ + s) * EDIM + dh;
  Qh[base]         = (_Float16)cq;
  Qh[base + DHEAD] = (_Float16)sq;
  Kh[base]         = (_Float16)ck;
  Kh[base + DHEAD] = (_Float16)sk;
}

// ---------------------------------------------------------------------------
// Kernel 2: V = x @ v_w^T, stored TRANSPOSED as Vt (B,H,DHEAD,S) f16 so that
// the attention kernel's B-fragments (columns over seq) are contiguous.
// Block = 4 waves, C tile 128x64, B tile async-staged through LDS.
// grid = (M/128)*(N/64).
// ---------------------------------------------------------------------------
__global__ void gemm_v_kernel(const _Float16* __restrict__ Xh,
                              const _Float16* __restrict__ Wv,
                              _Float16* __restrict__ Vt) {
  __shared__ __align__(64) _Float16 Bt[2 * 64 * 32];
  const int tid  = threadIdx.x;
  const int lane = tid & 31;
  const int wv   = tid >> 5;
  const int tile = blockIdx.x;
  const int mt = tile & ((BATCH * SEQ / 128) - 1);  // 32 m-tiles of 128
  const int ng = tile >> 5;                         // 16 n-groups of 64
  const int m0 = mt * 128 + wv * 32;
  const int n0 = ng * 64;

  v8f acc[8] = {};
  gemm_core(Xh, Wv, Bt, m0, n0, tid, acc);

  const int rowoff = (lane & 16) ? 8 : 0;
#pragma unroll
  for (int mi = 0; mi < 2; ++mi) {
#pragma unroll
    for (int j = 0; j < 4; ++j) {
      const int n = n0 + j * 16 + (lane & 15);
      const int h = n >> 6, dd = n & 63;
#pragma unroll
      for (int r = 0; r < 8; ++r) {
        const int m = m0 + mi * 16 + r + rowoff;
        const int bb = m >> 11, s = m & (SEQ - 1);
        Vt[(((size_t)(bb * NHEADS + h) * DHEAD + dd) << 11) + s] =
            (_Float16)acc[mi * 4 + j][r];
      }
    }
  }
}

// ---------------------------------------------------------------------------
// Kernel 3: causal flash attention. One wave per (b,h, 16-query tile).
// Scores via 8 WMMA per 32-key block; online softmax with __shfl_xor row
// reductions; P staged through LDS to re-layout into an A-fragment; P@V via
// 4 WMMA into f32 accumulators. Output Oh (B*S, DMODEL) f16.
// ---------------------------------------------------------------------------
__global__ void flash_kernel(const _Float16* __restrict__ Qh,
                             const _Float16* __restrict__ Kh,
                             const _Float16* __restrict__ Vt,
                             _Float16* __restrict__ Oh) {
  __shared__ __align__(64) _Float16 psh[16 * 32];

  const int lane = threadIdx.x;
  const int blk  = blockIdx.x;              // b*H*(S/16)
  const int qt   = blk & ((SEQ / 16) - 1);  // 128 query tiles
  const int bh   = blk >> 7;                // 0..31
  const int q0   = qt * 16;

  const _Float16* Qbh = Qh + (size_t)bh * SEQ * EDIM;
  const _Float16* Kbh = Kh + (size_t)bh * SEQ * EDIM;
  const _Float16* Vbh = Vt + (size_t)bh * DHEAD * SEQ;

  // Preload Q A-fragments for all 4 K-steps (128 features)
  v16h qa[4];
#pragma unroll
  for (int ks = 0; ks < 4; ++ks) qa[ks] = load_a_frag(Qbh, EDIM, q0, ks * 32, lane);

  v8f o[4] = {};
  float m_r[8], l_r[8];
#pragma unroll
  for (int r = 0; r < 8; ++r) { m_r[r] = -__builtin_inff(); l_r[r] = 0.0f; }

  const float scale  = 0.0883883476f;   // 1/sqrt(2*DHEAD)
  const int   ncol   = lane & 15;
  const int   rowoff = (lane & 16) ? 8 : 0;
  const int   kb_max = (q0 + 15) >> 5;

  for (int kb = 0; kb <= kb_max; ++kb) {
    const int kbase = kb * 32;

    // ---- scores: 16 queries x 32 keys ----
    v8f s0 = {}, s1 = {};
#pragma unroll
    for (int ks = 0; ks < 4; ++ks) {
      v16h b0 = load_b_frag(Kbh, EDIM, kbase,      ks * 32, lane);
      v16h b1 = load_b_frag(Kbh, EDIM, kbase + 16, ks * 32, lane);
      s0 = wmma16(qa[ks], b0, s0);
      s1 = wmma16(qa[ks], b1, s1);
    }

    // ---- scale + causal mask ----
#pragma unroll
    for (int r = 0; r < 8; ++r) {
      const int q = q0 + r + rowoff;
      float v0 = s0[r] * scale;
      float v1 = s1[r] * scale;
      if (kbase + ncol      > q) v0 = -__builtin_inff();
      if (kbase + 16 + ncol > q) v1 = -__builtin_inff();
      s0[r] = v0; s1[r] = v1;
    }

    // ---- online softmax stats (rows live in one VGPR across a 16-lane group)
    float alpha[8];
#pragma unroll
    for (int r = 0; r < 8; ++r) {
      float v = fmaxf(s0[r], s1[r]);
#pragma unroll
      for (int off = 8; off >= 1; off >>= 1)
        v = fmaxf(v, __shfl_xor(v, off, 32));
      const float mn = fmaxf(m_r[r], v);
      alpha[r] = __expf(m_r[r] - mn);
      m_r[r] = mn;
    }

#pragma unroll
    for (int r = 0; r < 8; ++r) {
      const float p0 = __expf(s0[r] - m_r[r]);
      const float p1 = __expf(s1[r] - m_r[r]);
      s0[r] = p0; s1[r] = p1;
      float ps = p0 + p1;
#pragma unroll
      for (int off = 8; off >= 1; off >>= 1)
        ps += __shfl_xor(ps, off, 32);
      l_r[r] = l_r[r] * alpha[r] + ps;
    }

#pragma unroll
    for (int j = 0; j < 4; ++j)
#pragma unroll
      for (int r = 0; r < 8; ++r) o[j][r] *= alpha[r];

    // ---- P: C-layout -> A-fragment via LDS (single wave, in-order DS) ----
#pragma unroll
    for (int r = 0; r < 8; ++r) {
      psh[(r + rowoff) * 32 + ncol]      = (_Float16)s0[r];
      psh[(r + rowoff) * 32 + 16 + ncol] = (_Float16)s1[r];
    }
    const _Float16* pr = psh + (lane & 15) * 32 + ((lane & 16) ? 16 : 0);
    v8h lo = *(const v8h*)(pr);
    v8h hi = *(const v8h*)(pr + 8);
    v16h pa = __builtin_shufflevector(lo, hi, 0,1,2,3,4,5,6,7,8,9,10,11,12,13,14,15);

    // ---- O += P @ V (B columns contiguous thanks to Vt layout) ----
#pragma unroll
    for (int j = 0; j < 4; ++j) {
      v16h vb = load_b_frag(Vbh, SEQ, j * 16, kbase, lane);
      o[j] = wmma16(pa, vb, o[j]);
    }
  }

  // ---- normalize + store O as f16 rows of (B*S, DMODEL) ----
  const int b = bh >> 4, h = bh & 15;
  float invl[8];
#pragma unroll
  for (int r = 0; r < 8; ++r) invl[r] = 1.0f / l_r[r];
#pragma unroll
  for (int j = 0; j < 4; ++j) {
    const int col = h * DHEAD + j * 16 + (lane & 15);
#pragma unroll
    for (int r = 0; r < 8; ++r) {
      const int q = q0 + r + rowoff;
      Oh[(((size_t)(b * SEQ + q)) << 10) + col] = (_Float16)(o[j][r] * invl[r]);
    }
  }
}

// ---------------------------------------------------------------------------
// Kernel 4: out = O @ out_w^T, f32 output. Same 4-wave async-staged scheme.
// ---------------------------------------------------------------------------
__global__ void gemm_out_kernel(const _Float16* __restrict__ Oh,
                                const _Float16* __restrict__ Wo,
                                float* __restrict__ out) {
  __shared__ __align__(64) _Float16 Bt[2 * 64 * 32];
  const int tid  = threadIdx.x;
  const int lane = tid & 31;
  const int wv   = tid >> 5;
  const int tile = blockIdx.x;
  const int mt = tile & ((BATCH * SEQ / 128) - 1);
  const int ng = tile >> 5;
  const int m0 = mt * 128 + wv * 32;
  const int n0 = ng * 64;

  v8f acc[8] = {};
  gemm_core(Oh, Wo, Bt, m0, n0, tid, acc);

  const int rowoff = (lane & 16) ? 8 : 0;
#pragma unroll
  for (int mi = 0; mi < 2; ++mi) {
#pragma unroll
    for (int j = 0; j < 4; ++j) {
      const int n = n0 + j * 16 + (lane & 15);
#pragma unroll
      for (int r = 0; r < 8; ++r) {
        const int m = m0 + mi * 16 + r + rowoff;
        out[((size_t)m << 10) + n] = acc[mi * 4 + j][r];
      }
    }
  }
}

// ---------------------------------------------------------------------------
// Host launcher
// ---------------------------------------------------------------------------
extern "C" void kernel_launch(void* const* d_in, const int* in_sizes, int n_in,
                              void* d_out, int out_size, void* d_ws, size_t ws_size,
                              hipStream_t stream) {
  const float* x     = (const float*)d_in[0];
  const float* w_q   = (const float*)d_in[1];
  const float* b_q   = (const float*)d_in[2];
  const float* w_k   = (const float*)d_in[3];
  const float* b_k   = (const float*)d_in[4];
  const float* v_w   = (const float*)d_in[5];
  const float* out_w = (const float*)d_in[6];
  float* out = (float*)d_out;

  char* p = (char*)d_ws;
  _Float16* Xh = (_Float16*)p; p += (size_t)BATCH * SEQ * DMODEL * 2;          // 8 MB
  _Float16* Wv = (_Float16*)p; p += (size_t)DMODEL * DMODEL * 2;               // 2 MB
  _Float16* Wo = (_Float16*)p; p += (size_t)DMODEL * DMODEL * 2;               // 2 MB
  _Float16* Qh = (_Float16*)p; p += (size_t)BATCH * NHEADS * SEQ * EDIM * 2;   // 16 MB
  _Float16* Kh = (_Float16*)p; p += (size_t)BATCH * NHEADS * SEQ * EDIM * 2;   // 16 MB
  _Float16* Vt = (_Float16*)p; p += (size_t)BATCH * NHEADS * DHEAD * SEQ * 2;  // 8 MB
  _Float16* Oh = (_Float16*)p; p += (size_t)BATCH * SEQ * DMODEL * 2;          // 8 MB

  f32_to_f16_kernel<<<DMODEL * DMODEL / 256, 256, 0, stream>>>(v_w, Wv);
  f32_to_f16_kernel<<<DMODEL * DMODEL / 256, 256, 0, stream>>>(out_w, Wo);
  qk_prep_kernel<<<BATCH * SEQ * DMODEL / 256, 256, 0, stream>>>(
      x, w_q, b_q, w_k, b_k, Xh, Qh, Kh);
  gemm_v_kernel<<<(BATCH * SEQ / 128) * (DMODEL / 64), 128, 0, stream>>>(Xh, Wv, Vt);
  flash_kernel<<<BATCH * NHEADS * (SEQ / 16), 32, 0, stream>>>(Qh, Kh, Vt, Oh);
  gemm_out_kernel<<<(BATCH * SEQ / 128) * (DMODEL / 64), 128, 0, stream>>>(Oh, Wo, out);
}
